// SemanticAlignmentFusion_v2_27522150432902
// MI455X (gfx1250) — compile-verified, matrix operand order checked by reading, not run on hardware
//
#include <hip/hip_runtime.h>

#define GNUM 3
#define C_ALL 768
#define GC 256
#define BATCH 8
#define LN_EPS 1e-5f

typedef __attribute__((ext_vector_type(16))) _Float16 v16h;
typedef __attribute__((ext_vector_type(8)))  float    v8f;

// ---------------------------------------------------------------------------
// Pass 1: logits[b,g,n] = b_qk + sum_c x[b, src(g*256+c), n] * w_qk[c]
// One thread per (b,n); loops all 768 channels (each input element read once).
// ---------------------------------------------------------------------------
__global__ void logits_kernel(const float* __restrict__ x,
                              const float* __restrict__ w_qk,
                              const float* __restrict__ b_qk,
                              float* __restrict__ logits, int N) {
  __shared__ float wq_s[GC];
  __shared__ int   cc_s[C_ALL];
  for (int i = threadIdx.x; i < C_ALL; i += blockDim.x) {
    if (i < GC) wq_s[i] = w_qk[i];
    cc_s[i] = (i % 3) * GC + i / 3;   // src channel for shuffled index j=i
  }
  __syncthreads();
  const int nt = N / 256;
  const int b  = blockIdx.x / nt;
  const int n  = (blockIdx.x % nt) * 256 + threadIdx.x;
  const float* xb = x + (size_t)b * C_ALL * N + n;
  const float bq = b_qk[0];
  for (int g = 0; g < GNUM; ++g) {
    float acc = 0.f;
#pragma unroll 4
    for (int c = 0; c < GC; ++c) {
      const int cc = cc_s[g * GC + c];
      acc = fmaf(xb[(size_t)cc * N], wq_s[c], acc);
    }
    logits[((size_t)b * GNUM + g) * N + n] = acc + bq;
  }
}

// ---------------------------------------------------------------------------
// Softmax stats per (b,g): rowmax and sum(exp(l - max)) over N.
// ---------------------------------------------------------------------------
__global__ void softmax_stats_kernel(const float* __restrict__ logits,
                                     float* __restrict__ stats, int N) {
  const int bg = blockIdx.x;                  // 0..23
  const float* l = logits + (size_t)bg * N;
  __shared__ float red[256];
  float m = -3.4e38f;
  for (int n = threadIdx.x; n < N; n += 256) m = fmaxf(m, l[n]);
  red[threadIdx.x] = m; __syncthreads();
  for (int s = 128; s > 0; s >>= 1) {
    if (threadIdx.x < s) red[threadIdx.x] = fmaxf(red[threadIdx.x], red[threadIdx.x + s]);
    __syncthreads();
  }
  m = red[0]; __syncthreads();
  float sum = 0.f;
  for (int n = threadIdx.x; n < N; n += 256) sum += expf(l[n] - m);
  red[threadIdx.x] = sum; __syncthreads();
  for (int s = 128; s > 0; s >>= 1) {
    if (threadIdx.x < s) red[threadIdx.x] += red[threadIdx.x + s];
    __syncthreads();
  }
  if (threadIdx.x == 0) { stats[2 * bg] = m; stats[2 * bg + 1] = red[0]; }
}

// ---------------------------------------------------------------------------
// Pass 2: context[b, j] = sum_n x[b,cc,n] * attn[b,g,n], j = shuffled index.
// One block per (b, cc); strided reduce over n. Emits global_prefetch_b8.
// ---------------------------------------------------------------------------
__global__ void context_kernel(const float* __restrict__ x,
                               const float* __restrict__ logits,
                               const float* __restrict__ stats,
                               float* __restrict__ context, int N) {
  const int cc = blockIdx.x % C_ALL;
  const int b  = blockIdx.x / C_ALL;
  const int j  = (cc & 255) * 3 + (cc >> 8);  // g*256 + c
  const int g  = j >> 8;
  const float* xrow = x + ((size_t)b * C_ALL + cc) * N;
  const float* lrow = logits + ((size_t)b * GNUM + g) * N;
  const float m   = stats[2 * (b * GNUM + g)];
  const float inv = 1.f / stats[2 * (b * GNUM + g) + 1];
  float acc = 0.f;
  for (int n = threadIdx.x; n < N; n += 256) {
    __builtin_prefetch(xrow + n + 4096, 0, 1);   // global_prefetch_b8
    acc = fmaf(xrow[n], expf(lrow[n] - m), acc);
  }
  __shared__ float red[256];
  red[threadIdx.x] = acc; __syncthreads();
  for (int s = 128; s > 0; s >>= 1) {
    if (threadIdx.x < s) red[threadIdx.x] += red[threadIdx.x + s];
    __syncthreads();
  }
  if (threadIdx.x == 0) context[(size_t)b * C_ALL + j] = red[0] * inv;
}

// ---------------------------------------------------------------------------
// ctx_transform: per (b,g) row: 256 -> 64 (LN, ReLU) -> 256.
// ---------------------------------------------------------------------------
__global__ void ct_transform_kernel(const float* __restrict__ context,
                                    const float* __restrict__ w1, const float* __restrict__ b1,
                                    const float* __restrict__ gam, const float* __restrict__ bet,
                                    const float* __restrict__ w2, const float* __restrict__ b2,
                                    float* __restrict__ ctx2) {
  const int bg = blockIdx.x;                  // b*3+g
  const int t  = threadIdx.x;                 // 0..63
  const int b  = bg / GNUM, g = bg % GNUM;
  const float* row = context + (size_t)b * C_ALL + g * GC;
  float h = b1[t];
  for (int c = 0; c < GC; ++c) h = fmaf(row[c], w1[t * GC + c], h);
  __shared__ float r1[64], r2[64], hr[64];
  r1[t] = h; r2[t] = h * h; __syncthreads();
  for (int s = 32; s > 0; s >>= 1) {
    if (t < s) { r1[t] += r1[t + s]; r2[t] += r2[t + s]; }
    __syncthreads();
  }
  const float mean = r1[0] * (1.f / 64.f);
  const float var  = r2[0] * (1.f / 64.f) - mean * mean;
  const float rstd = rsqrtf(var + LN_EPS);
  hr[t] = fmaxf((h - mean) * rstd * gam[t] + bet[t], 0.f);
  __syncthreads();
  for (int o = t; o < GC; o += 64) {
    float acc = b2[o];
    for (int k = 0; k < 64; ++k) acc = fmaf(hr[k], w2[o * 64 + k], acc);
    ctx2[(size_t)b * C_ALL + g * GC + o] = acc;
  }
}

// ---------------------------------------------------------------------------
// lvl_weight first matmul via WMMA: y1[8,768] = context[8,768] @ w1[768,768]^T
// One wave per 16-wide N tile; M padded 8->16 (branchless row-clamp + mask);
// K = 24 steps of 32; b128 vector loads; fp16 operands, fp32 accumulation.
// ---------------------------------------------------------------------------
__global__ __launch_bounds__(32)
void lw1_wmma_kernel(const float* __restrict__ ctx,   // [8,768]
                     const float* __restrict__ w1,    // [768,768] row-major
                     const float* __restrict__ b1,    // [768]
                     float* __restrict__ y1) {        // [8,768]
  const int t        = blockIdx.x;          // N tile 0..47
  const int lane     = threadIdx.x;         // 0..31
  const int mrow     = lane & 15;           // A row / B,D column-within-tile
  const int half_sel = lane >> 4;           // 0 or 1
  const int kb       = half_sel * 8;        // K base for this half-wave
  const int n        = t * 16 + mrow;       // output column
  const float amask  = (mrow < BATCH) ? 1.f : 0.f;   // zero-pad rows 8..15
  const float* arow  = ctx + (size_t)(mrow & 7) * C_ALL;   // clamped, valid
  const float* wrow  = w1 + (size_t)n * C_ALL;
  v8f acc = {};
  for (int k0 = 0; k0 < C_ALL; k0 += 32) {
    const float4 a0 = *reinterpret_cast<const float4*>(arow + k0 + kb);
    const float4 a1 = *reinterpret_cast<const float4*>(arow + k0 + kb + 4);
    const float4 a2 = *reinterpret_cast<const float4*>(arow + k0 + kb + 16);
    const float4 a3 = *reinterpret_cast<const float4*>(arow + k0 + kb + 20);
    const float4 b0 = *reinterpret_cast<const float4*>(wrow + k0 + kb);
    const float4 b1v = *reinterpret_cast<const float4*>(wrow + k0 + kb + 4);
    const float4 b2 = *reinterpret_cast<const float4*>(wrow + k0 + kb + 16);
    const float4 b3 = *reinterpret_cast<const float4*>(wrow + k0 + kb + 20);
    v16h av, bv;
    av[0]  = (_Float16)(a0.x * amask); av[1]  = (_Float16)(a0.y * amask);
    av[2]  = (_Float16)(a0.z * amask); av[3]  = (_Float16)(a0.w * amask);
    av[4]  = (_Float16)(a1.x * amask); av[5]  = (_Float16)(a1.y * amask);
    av[6]  = (_Float16)(a1.z * amask); av[7]  = (_Float16)(a1.w * amask);
    av[8]  = (_Float16)(a2.x * amask); av[9]  = (_Float16)(a2.y * amask);
    av[10] = (_Float16)(a2.z * amask); av[11] = (_Float16)(a2.w * amask);
    av[12] = (_Float16)(a3.x * amask); av[13] = (_Float16)(a3.y * amask);
    av[14] = (_Float16)(a3.z * amask); av[15] = (_Float16)(a3.w * amask);
    bv[0]  = (_Float16)b0.x;  bv[1]  = (_Float16)b0.y;
    bv[2]  = (_Float16)b0.z;  bv[3]  = (_Float16)b0.w;
    bv[4]  = (_Float16)b1v.x; bv[5]  = (_Float16)b1v.y;
    bv[6]  = (_Float16)b1v.z; bv[7]  = (_Float16)b1v.w;
    bv[8]  = (_Float16)b2.x;  bv[9]  = (_Float16)b2.y;
    bv[10] = (_Float16)b2.z;  bv[11] = (_Float16)b2.w;
    bv[12] = (_Float16)b3.x;  bv[13] = (_Float16)b3.y;
    bv[14] = (_Float16)b3.z;  bv[15] = (_Float16)b3.w;
    acc = __builtin_amdgcn_wmma_f32_16x16x32_f16(false, av, false, bv,
                                                 (short)0, acc, false, false);
  }
#pragma unroll
  for (int r = 0; r < 8; ++r) {
    const int M = r + half_sel * 8;
    if (M < BATCH) y1[(size_t)M * C_ALL + n] = acc[r] + b1[n];
  }
}

// ---------------------------------------------------------------------------
// LayerNorm(768) + ReLU, in place, one block per batch row.
// ---------------------------------------------------------------------------
__global__ void ln_relu_768_kernel(float* __restrict__ y,
                                   const float* __restrict__ gam,
                                   const float* __restrict__ bet) {
  const int b = blockIdx.x;
  float* row = y + (size_t)b * C_ALL;
  __shared__ float r1[256], r2[256];
  float s1 = 0.f, s2 = 0.f;
  for (int k = threadIdx.x; k < C_ALL; k += 256) { float v = row[k]; s1 += v; s2 += v * v; }
  r1[threadIdx.x] = s1; r2[threadIdx.x] = s2; __syncthreads();
  for (int s = 128; s > 0; s >>= 1) {
    if (threadIdx.x < s) { r1[threadIdx.x] += r1[threadIdx.x + s]; r2[threadIdx.x] += r2[threadIdx.x + s]; }
    __syncthreads();
  }
  const float mean = r1[0] * (1.f / 768.f);
  const float var  = r2[0] * (1.f / 768.f) - mean * mean;
  const float rstd = rsqrtf(var + LN_EPS);
  for (int k = threadIdx.x; k < C_ALL; k += 256)
    row[k] = fmaxf((row[k] - mean) * rstd * gam[k] + bet[k], 0.f);
}

// lvlw[b,g] = sigmoid(dot(l1[b,:], lw_w2[g,:]) + lw_b2[g])
__global__ void lvlw_kernel(const float* __restrict__ l1,
                            const float* __restrict__ w2, const float* __restrict__ b2,
                            float* __restrict__ lvlw) {
  const int bg = blockIdx.x; const int b = bg / GNUM, g = bg % GNUM;
  const float* row = l1 + (size_t)b * C_ALL;
  const float* w   = w2 + (size_t)g * C_ALL;
  float acc = 0.f;
  for (int k = threadIdx.x; k < C_ALL; k += 256) acc = fmaf(row[k], w[k], acc);
  __shared__ float red[256];
  red[threadIdx.x] = acc; __syncthreads();
  for (int s = 128; s > 0; s >>= 1) {
    if (threadIdx.x < s) red[threadIdx.x] += red[threadIdx.x + s];
    __syncthreads();
  }
  if (threadIdx.x == 0) lvlw[bg] = 1.f / (1.f + expf(-(red[0] + b2[g])));
}

// ctxSum[b,c] = sum_g ctx2[b,g,c] * lvlw[b,g]
__global__ void ctxsum_kernel(const float* __restrict__ ctx2,
                              const float* __restrict__ lvlw,
                              float* __restrict__ ctxSum) {
  const int b = blockIdx.x, c = threadIdx.x;
  float acc = 0.f;
#pragma unroll
  for (int g = 0; g < GNUM; ++g)
    acc = fmaf(ctx2[(size_t)b * C_ALL + g * GC + c], lvlw[b * GNUM + g], acc);
  ctxSum[b * GC + c] = acc;
}

// ---------------------------------------------------------------------------
// Pass 3: out[b,c,n] = ctxSum[b,c] + sum_g x[b, src(g*256+c), n]
// ---------------------------------------------------------------------------
__global__ void output_kernel(const float* __restrict__ x,
                              const float* __restrict__ ctxSum,
                              float* __restrict__ out, int N) {
  const int nt  = N / 256;
  const int n   = (blockIdx.x % nt) * 256 + threadIdx.x;
  const int tmp = blockIdx.x / nt;
  const int c   = tmp % GC;
  const int b   = tmp / GC;
  const float* xb = x + (size_t)b * C_ALL * N;
  float acc = ctxSum[b * GC + c];
#pragma unroll
  for (int g = 0; g < GNUM; ++g) {
    const int j  = g * GC + c;
    const int cc = (j % 3) * GC + j / 3;
    acc += xb[(size_t)cc * N + n];
  }
  out[((size_t)b * GC + c) * N + n] = acc;
}

extern "C" void kernel_launch(void* const* d_in, const int* in_sizes, int n_in,
                              void* d_out, int out_size, void* d_ws, size_t ws_size,
                              hipStream_t stream) {
  const float* g[4]   = {(const float*)d_in[0], (const float*)d_in[1],
                         (const float*)d_in[2], (const float*)d_in[3]};
  const float* x_last = (const float*)d_in[4];
  const float* w_qk   = (const float*)d_in[5];
  const float* b_qk   = (const float*)d_in[6];
  const float* ct_w1  = (const float*)d_in[7];
  const float* ct_b1  = (const float*)d_in[8];
  const float* ct_g   = (const float*)d_in[9];
  const float* ct_be  = (const float*)d_in[10];
  const float* ct_w2  = (const float*)d_in[11];
  const float* ct_b2  = (const float*)d_in[12];
  const float* lw_w1  = (const float*)d_in[13];
  const float* lw_b1  = (const float*)d_in[14];
  const float* lw_g   = (const float*)d_in[15];
  const float* lw_be  = (const float*)d_in[16];
  const float* lw_w2  = (const float*)d_in[17];
  const float* lw_b2  = (const float*)d_in[18];

  float* out = (float*)d_out;
  float* ws  = (float*)d_ws;

  static const int    Ns[4]         = {16384, 4096, 1024, 256};
  static const size_t logits_off[4] = {0, 393216, 491520, 516096};
  static const size_t out_off[4]    = {0, 33554432, 41943040, 44040192};

  for (int lev = 0; lev < 4; ++lev) {
    const int N = Ns[lev];
    float* logits  = ws + logits_off[lev];
    float* stats   = ws + 522240 + (size_t)lev * 48;
    float* context = ws + 522432 + (size_t)lev * 6144;
    float* ctx2    = ws + 547008 + (size_t)lev * 6144;
    float* y1      = ws + 571584 + (size_t)lev * 6144;
    float* lvlw    = ws + 596160 + (size_t)lev * 24;
    float* csum    = ws + 596256 + (size_t)lev * 2048;
    const float* x = g[lev];

    logits_kernel<<<BATCH * (N / 256), 256, 0, stream>>>(
        x, w_qk + lev * GC, b_qk + lev, logits, N);
    softmax_stats_kernel<<<BATCH * GNUM, 256, 0, stream>>>(logits, stats, N);
    context_kernel<<<BATCH * C_ALL, 256, 0, stream>>>(x, logits, stats, context, N);
    ct_transform_kernel<<<BATCH * GNUM, 64, 0, stream>>>(
        context, ct_w1 + (size_t)lev * 64 * GC, ct_b1 + lev * 64,
        ct_g + lev * 64, ct_be + lev * 64,
        ct_w2 + (size_t)lev * GC * 64, ct_b2 + lev * GC, ctx2);
    lw1_wmma_kernel<<<C_ALL / 16, 32, 0, stream>>>(
        context, lw_w1 + (size_t)lev * C_ALL * C_ALL, lw_b1 + lev * C_ALL, y1);
    ln_relu_768_kernel<<<BATCH, 256, 0, stream>>>(y1, lw_g + lev * C_ALL, lw_be + lev * C_ALL);
    lvlw_kernel<<<BATCH * GNUM, 256, 0, stream>>>(
        y1, lw_w2 + (size_t)lev * GNUM * C_ALL, lw_b2 + lev * GNUM, lvlw);
    ctxsum_kernel<<<BATCH, 256, 0, stream>>>(ctx2, lvlw, csum);
    output_kernel<<<BATCH * GC * (N / 256), 256, 0, stream>>>(x, csum, out + out_off[lev], N);
  }
  // Pass-through x_last (capture-safe async D2D copy).
  hipMemcpyAsync(out + 44564480, x_last, (size_t)BATCH * 256 * 64 * sizeof(float),
                 hipMemcpyDeviceToDevice, stream);
}
